// LPN_48017734369829
// MI455X (gfx1250) — matrix operand order repaired and checked by reading, not compiled, not run on hardware
//
#include <hip/hip_runtime.h>

// MI455X / gfx1250: wave32, WMMA 16x16x32 bf16 with f32 accumulation.
// All heavy GEMMs use v_wmma_f32_16x16x32_bf16; score attention is a
// flash-style online softmax (sim is never materialized).
// Each GEMM wave carries 4 accumulator tiles so every A-fragment feeds
// 4 back-to-back v_wmma issues; f32 operands are loaded as b128.

typedef __attribute__((ext_vector_type(16))) __bf16 v16bf;
typedef __attribute__((ext_vector_type(8)))  float  v8f;

union FragU { v16bf v; uint4 q[2]; };

// ---- WMMA fragment loaders (contiguous k = hi*16 + s, same in A and B, wave32) ----------
// A 16x32 bf16: lane m = lane&15; VGPR0-7 K = hi*16 .. +15
__device__ __forceinline__ v16bf fragA_bf16(const __bf16* M, int ld, int row, int hi, int k0) {
    const __bf16* p = M + (size_t)row * ld + k0 + hi * 16;
    FragU u;
    u.q[0] = *(const uint4*)p;
    u.q[1] = *(const uint4*)(p + 8);
    return u.v;
}
// B 32x16 bf16 (computing A @ Bm^T, Bm row-major): lane n = lane&15 holds
// Bm[n][k0 + hi*16 .. +15] (contiguous 32 bytes).
__device__ __forceinline__ v16bf fragB_bf16(const __bf16* M, int ld, int row, int hi, int k0) {
    const __bf16* p = M + (size_t)row * ld + k0 + hi * 16;
    FragU u;
    u.q[0] = ((const uint4*)p)[0];
    u.q[1] = ((const uint4*)p)[1];
    return u.v;
}
// f32 sources: b128 loads + RNE f32->bf16 (v_cvt_pk_bf16_f32).
__device__ __forceinline__ v16bf fragA_f32(const float* M, int ld, int row, int hi, int k0) {
    const float* p = M + (size_t)row * ld + k0 + hi * 16;   // 64B aligned
    float4 a0 = ((const float4*)p)[0];
    float4 a1 = ((const float4*)p)[1];
    float4 a2 = ((const float4*)(p + 8))[0];
    float4 a3 = ((const float4*)(p + 8))[1];
    v16bf v;
    v[0]  = (__bf16)a0.x; v[1]  = (__bf16)a0.y; v[2]  = (__bf16)a0.z; v[3]  = (__bf16)a0.w;
    v[4]  = (__bf16)a1.x; v[5]  = (__bf16)a1.y; v[6]  = (__bf16)a1.z; v[7]  = (__bf16)a1.w;
    v[8]  = (__bf16)a2.x; v[9]  = (__bf16)a2.y; v[10] = (__bf16)a2.z; v[11] = (__bf16)a2.w;
    v[12] = (__bf16)a3.x; v[13] = (__bf16)a3.y; v[14] = (__bf16)a3.z; v[15] = (__bf16)a3.w;
    return v;
}
__device__ __forceinline__ v16bf fragB_f32(const float* M, int ld, int row, int hi, int k0) {
    const float* p = M + (size_t)row * ld + k0 + hi * 16;   // 64B aligned
    v16bf v;
#pragma unroll
    for (int q = 0; q < 4; ++q) {
        float4 a = ((const float4*)p)[q];
        v[4 * q]     = (__bf16)a.x;
        v[4 * q + 1] = (__bf16)a.y;
        v[4 * q + 2] = (__bf16)a.z;
        v[4 * q + 3] = (__bf16)a.w;
    }
    return v;
}
__device__ __forceinline__ v8f wmma_bf16(v16bf a, v16bf b, v8f c) {
    return __builtin_amdgcn_wmma_f32_16x16x32_bf16(false, a, false, b, (short)0, c, false, false);
}
// Half-wave (16-lane) reductions: D-tile rows live across lanes 0-15 / 16-31.
__device__ __forceinline__ float halfmax(float v) {
    v = fmaxf(v, __shfl_xor(v, 1, 32)); v = fmaxf(v, __shfl_xor(v, 2, 32));
    v = fmaxf(v, __shfl_xor(v, 4, 32)); v = fmaxf(v, __shfl_xor(v, 8, 32));
    return v;
}
__device__ __forceinline__ float halfsum(float v) {
    v += __shfl_xor(v, 1, 32); v += __shfl_xor(v, 2, 32);
    v += __shfl_xor(v, 4, 32); v += __shfl_xor(v, 8, 32);
    return v;
}

// ---------------------------------------------------------------------------
// K1: feat_total[b] = [concat(feat_nor,feat_abn)@Wq^T+bq ; feat_hard@Wk^T+bk]
//     Also writes row-L2-normalized bf16 copy (featn) for the attention GEMMs.
//     4 waves/block; each wave: 16-row strip x full 1024 outputs,
//     4 N-tiles in flight per A fragment.
__global__ void k_proj(const float* __restrict__ feat_nor, const float* __restrict__ feat_abn,
                       const float* __restrict__ feat_hard,
                       const float* __restrict__ Wq, const float* __restrict__ bq,
                       const float* __restrict__ Wk, const float* __restrict__ bk,
                       float* __restrict__ feat_total, __bf16* __restrict__ featn) {
    const int wid = threadIdx.x >> 5, lane = threadIdx.x & 31;
    const int c = lane & 15, hi = lane >> 4;
    const int strip = (blockIdx.x * 4 + wid) * 16;   // 0..2032
    const int b = blockIdx.y;
    const float *X, *W, *bias; int xrow;
    if (strip < 512)       { X = feat_nor  + (size_t)b * 512 * 1024;  xrow = strip;        W = Wq; bias = bq; }
    else if (strip < 1024) { X = feat_abn  + (size_t)b * 512 * 1024;  xrow = strip - 512;  W = Wq; bias = bq; }
    else                   { X = feat_hard + (size_t)b * 1024 * 1024; xrow = strip - 1024; W = Wk; bias = bk; }
    float* orow = feat_total + ((size_t)b * 2048 + strip) * 1024;
    const float* Xs = X + (size_t)xrow * 1024;

    float ss[8];
#pragma unroll
    for (int i = 0; i < 8; ++i) ss[i] = 0.f;

    for (int ng = 0; ng < 16; ++ng) {                 // 4 N-tiles per group
        v8f acc[4] = {{}, {}, {}, {}};
        for (int kt = 0; kt < 1024; kt += 32) {
            v16bf a = fragA_f32(Xs, 1024, c, hi, kt);
#pragma unroll
            for (int q = 0; q < 4; ++q) {
                v16bf bb = fragB_f32(W, 1024, ng * 64 + q * 16 + c, hi, kt);
                acc[q] = wmma_bf16(a, bb, acc[q]);
            }
        }
#pragma unroll
        for (int q = 0; q < 4; ++q) {
            const int f = ng * 64 + q * 16 + c;
            const float bv = bias[f];
#pragma unroll
            for (int i = 0; i < 8; ++i) {
                float v = acc[q][i] + bv;
                orow[(size_t)(i + 8 * hi) * 1024 + f] = v;
                ss[i] += v * v;
            }
        }
    }
    float invn[8];
#pragma unroll
    for (int i = 0; i < 8; ++i) invn[i] = rsqrtf(halfsum(ss[i]));
    __bf16* nrow = featn + ((size_t)b * 2048 + strip) * 1024;
    for (int nt = 0; nt < 64; ++nt) {
        const int f = nt * 16 + c;
#pragma unroll
        for (int i = 0; i < 8; ++i) {
            size_t off = (size_t)(i + 8 * hi) * 1024 + f;
            nrow[off] = (__bf16)(orow[off] * invn[i]);      // same-wave store->load, in-order
        }
    }
}

// ---------------------------------------------------------------------------
// K2: new_score_hard[b,n] = softmax_m( fhn[n]·featn[m] ) · score[m]
//     Flash-style: online (max, denom, weighted-sum) per row; sim never stored.
//     4 M-tiles in flight per A fragment.
__global__ void k_attn_score(const __bf16* __restrict__ featn,
                             const float* __restrict__ score_abn,
                             float* __restrict__ new_score) {
    const int wid = threadIdx.x >> 5, lane = threadIdx.x & 31;
    const int c = lane & 15, hi = lane >> 4;
    const int strip = (blockIdx.x * 4 + wid) * 16;   // fh row strip (0..1008)
    const int b = blockIdx.y;
    const __bf16* fn  = featn + (size_t)b * 2048 * 1024;          // normalized feat
    const __bf16* fhn = fn + (size_t)1024 * 1024 + (size_t)strip * 1024; // normalized fh strip
    const float* sab = score_abn + (size_t)b * 512;

    float rmax[8], rden[8], rw[8];
#pragma unroll
    for (int i = 0; i < 8; ++i) { rmax[i] = -__builtin_inff(); rden[i] = 0.f; rw[i] = 0.f; }

    for (int mg = 0; mg < 16; ++mg) {                 // 4 M-tiles per group
        v8f acc[4] = {{}, {}, {}, {}};
        for (int kt = 0; kt < 1024; kt += 32) {
            v16bf a = fragA_bf16(fhn, 1024, c, hi, kt);
#pragma unroll
            for (int q = 0; q < 4; ++q) {
                v16bf bb = fragB_bf16(fn, 1024, mg * 64 + q * 16 + c, hi, kt);
                acc[q] = wmma_bf16(a, bb, acc[q]);
            }
        }
#pragma unroll
        for (int q = 0; q < 4; ++q) {
            const int m = mg * 64 + q * 16 + c;
            const float sm = (m < 512) ? 0.f : sab[m - 512];   // score = [0; score_conf_abn]
#pragma unroll
            for (int i = 0; i < 8; ++i) {
                float s    = acc[q][i];
                float nmax = fmaxf(rmax[i], halfmax(s));
                float scale = __expf(rmax[i] - nmax);          // 0 on first tile (-inf)
                float e     = __expf(s - nmax);
                rden[i] = rden[i] * scale + halfsum(e);
                rw[i]   = rw[i]   * scale + halfsum(e * sm);
                rmax[i] = nmax;
            }
        }
    }
    if (c == 0) {
#pragma unroll
        for (int i = 0; i < 8; ++i)
            new_score[(size_t)b * 1024 + strip + i + 8 * hi] = rw[i] / rden[i];
    }
}

// ---------------------------------------------------------------------------
// K3: per-row L2 normalize (f32 in -> bf16 out), one block per row.
__global__ void k_rownorm(const float* __restrict__ in, __bf16* __restrict__ out, int F) {
    const size_t row = blockIdx.x;
    const float* p = in + row * (size_t)F;
    __shared__ float red[256];
    const int tid = threadIdx.x;
    float ss = 0.f;
    for (int i = tid; i < F; i += 256) { float v = p[i]; ss += v * v; }
    red[tid] = ss; __syncthreads();
    for (int s = 128; s > 0; s >>= 1) { if (tid < s) red[tid] += red[tid + s]; __syncthreads(); }
    const float inv = rsqrtf(red[0]);
    __bf16* o = out + row * (size_t)F;
    for (int i = tid; i < F; i += 256) o[i] = (__bf16)(p[i] * inv);
}

// ---------------------------------------------------------------------------
// K4: S[b] = xn[b] @ xn[b]^T  (512x512, K=1024).  4 waves/block; each wave
//     owns one 16-row strip x one 64-col group (4 accumulator tiles).
__global__ void k_gemm_nt_s(const __bf16* __restrict__ xn, float* __restrict__ S) {
    const int wid = threadIdx.x >> 5, lane = threadIdx.x & 31;
    const int c = lane & 15, hi = lane >> 4;
    const int tile = blockIdx.x * 4 + wid;           // 0..255 (32 i-tiles x 8 j-groups)
    const int i0 = (tile >> 3) * 16, jg = (tile & 7) * 64;
    const int b = blockIdx.y;
    const __bf16* X = xn + (size_t)b * 512 * 1024;
    v8f acc[4] = {{}, {}, {}, {}};
    for (int kt = 0; kt < 1024; kt += 32) {
        v16bf a = fragA_bf16(X + (size_t)i0 * 1024, 1024, c, hi, kt);
#pragma unroll
        for (int q = 0; q < 4; ++q) {
            v16bf bb = fragB_bf16(X, 1024, jg + q * 16 + c, hi, kt);
            acc[q] = wmma_bf16(a, bb, acc[q]);
        }
    }
    float* Sb = S + (size_t)b * 512 * 512;
#pragma unroll
    for (int q = 0; q < 4; ++q)
#pragma unroll
        for (int i = 0; i < 8; ++i)
            Sb[(size_t)(i0 + i + 8 * hi) * 512 + jg + q * 16 + c] = acc[q][i];
}

// ---------------------------------------------------------------------------
// K5: row softmax of S (f32) -> P (bf16). One block per row of 512.
__global__ void k_softmax_p(const float* __restrict__ S, __bf16* __restrict__ P) {
    const size_t row = blockIdx.x;
    const float* s = S + row * 512;
    __shared__ float red[256];
    const int tid = threadIdx.x;
    float m = -__builtin_inff();
    for (int i = tid; i < 512; i += 256) m = fmaxf(m, s[i]);
    red[tid] = m; __syncthreads();
    for (int st = 128; st > 0; st >>= 1) { if (tid < st) red[tid] = fmaxf(red[tid], red[tid + st]); __syncthreads(); }
    m = red[0]; __syncthreads();
    float sum = 0.f;
    for (int i = tid; i < 512; i += 256) sum += __expf(s[i] - m);
    red[tid] = sum; __syncthreads();
    for (int st = 128; st > 0; st >>= 1) { if (tid < st) red[tid] += red[tid + st]; __syncthreads(); }
    const float inv = 1.f / red[0];
    __bf16* p = P + row * 512;
    for (int i = tid; i < 512; i += 256) p[i] = (__bf16)(__expf(s[i] - m) * inv);
}

// ---------------------------------------------------------------------------
// K6: C[b] = P[b](512x512,bf16) @ V[b](512x1024,f32).  V is NOT transposed in
// memory, so stage 32x64 V chunks into LDS as transposed bf16; 4 waves share
// the tile, each wave owns a 16-column slice. B-frags come from ds_load_b128.
__global__ void k_gemm_pv(const __bf16* __restrict__ P, const float* __restrict__ V,
                          float* __restrict__ C) {
    const int b  = blockIdx.z;
    const int i0 = blockIdx.x * 16;     // row tile
    const int n0 = blockIdx.y * 64;     // col tile (block)
    const int tid = threadIdx.x, wid = tid >> 5, lane = tid & 31;
    const int c = lane & 15, hi = lane >> 4;
    const __bf16* Pb = P + (size_t)b * 512 * 512 + (size_t)i0 * 512;
    const float*  Vb = V + (size_t)b * 512 * 1024;
    __shared__ __align__(16) __bf16 Vt[64][40];   // [n_local][k_local], +8 pad
    v8f acc = {};
    for (int kt = 0; kt < 512; kt += 32) {
        for (int e = tid; e < 2048; e += 128) {   // 32(K) x 64(N), coalesced on N
            int kk = e >> 6, nn = e & 63;
            Vt[nn][kk] = (__bf16)Vb[(size_t)(kt + kk) * 1024 + n0 + nn];
        }
        __syncthreads();
        v16bf a = fragA_bf16(Pb, 512, c, hi, kt);
        const __bf16* bp = &Vt[wid * 16 + c][hi * 16];
        FragU u; u.q[0] = ((const uint4*)bp)[0]; u.q[1] = ((const uint4*)bp)[1];
        acc = wmma_bf16(a, u.v, acc);
        __syncthreads();
    }
#pragma unroll
    for (int i = 0; i < 8; ++i)
        C[((size_t)b * 512 + i0 + i + 8 * hi) * 1024 + n0 + wid * 16 + c] = acc[i];
}

// ---------------------------------------------------------------------------
// K7: top-k selections (stable: ties -> lower index, matching lax.top_k) and
//     feature gathers. One block per batch.
__global__ void k_topk_gather(const float* __restrict__ score_abn,
                              const float* __restrict__ new_score,
                              const float* __restrict__ feat_total,
                              float* __restrict__ ftop_nor, float* __restrict__ ftop_abn,
                              float* __restrict__ out_sc_nor, float* __restrict__ out_sc_abn) {
    const int b = blockIdx.x, tid = threadIdx.x;
    __shared__ float sc[2048];
    __shared__ unsigned char used[2048];
    __shared__ float rv[256]; __shared__ int ri[256];
    __shared__ int sel[10];
    for (int i = tid; i < 2048; i += 256) {
        float v;
        if (i < 512)       v = 0.f;
        else if (i < 1024) v = score_abn[(size_t)b * 512 + i - 512];
        else               v = new_score[(size_t)b * 1024 + i - 1024];
        sc[i] = v; used[i] = 0;
    }
    __syncthreads();

#define SELECT(KMAX, LO, IS_MIN, BODY)                                              \
    for (int k = 0; k < (KMAX); ++k) {                                              \
        float bv = (IS_MIN) ? __builtin_inff() : -__builtin_inff(); int bi = 1 << 30; \
        for (int i = (LO) + tid; i < 2048; i += 256) if (!used[i]) {                \
            float v = sc[i];                                                        \
            bool better = (IS_MIN) ? (v < bv || (v == bv && i < bi))                \
                                   : (v > bv || (v == bv && i < bi));               \
            if (better) { bv = v; bi = i; }                                         \
        }                                                                           \
        rv[tid] = bv; ri[tid] = bi; __syncthreads();                                \
        for (int s = 128; s > 0; s >>= 1) {                                         \
            if (tid < s) {                                                          \
                float ov = rv[tid + s]; int oi = ri[tid + s];                       \
                bool better = (IS_MIN) ? (ov < rv[tid] || (ov == rv[tid] && oi < ri[tid])) \
                                       : (ov > rv[tid] || (ov == rv[tid] && oi < ri[tid])); \
                if (better) { rv[tid] = ov; ri[tid] = oi; }                         \
            }                                                                       \
            __syncthreads();                                                        \
        }                                                                           \
        if (tid == 0) { used[ri[0]] = 1; BODY; }                                    \
        __syncthreads();                                                            \
    }
#define RESET_USED for (int i = tid; i < 2048; i += 256) used[i] = 0; __syncthreads();

    // bottom-10 indices over score_total -> gather feat rows
    SELECT(10, 0, true, sel[k] = ri[0]);
    for (int k = 0; k < 10; ++k) {
        const float* src = feat_total + ((size_t)b * 2048 + sel[k]) * 1024;
        float* dst = ftop_nor + ((size_t)b * 10 + k) * 1024;
        for (int f = tid; f < 1024; f += 256) dst[f] = src[f];
    }
    __syncthreads();
    RESET_USED;
    // bottom-5 VALUES of new_score_hard (ascending, = -top_k(-x))
    SELECT(5, 1024, true, out_sc_nor[(size_t)b * 5 + k] = rv[0]);
    RESET_USED;
    // top-2 VALUES of new_score_hard (descending)
    SELECT(2, 1024, false, out_sc_abn[(size_t)b * 2 + k] = rv[0]);
    RESET_USED;
    // top-5 indices over score_total -> gather feat rows
    SELECT(5, 0, false, sel[k] = ri[0]);
    for (int k = 0; k < 5; ++k) {
        const float* src = feat_total + ((size_t)b * 2048 + sel[k]) * 1024;
        float* dst = ftop_abn + ((size_t)b * 5 + k) * 1024;
        for (int f = tid; f < 1024; f += 256) dst[f] = src[f];
    }
#undef SELECT
#undef RESET_USED
}

// ---------------------------------------------------------------------------
// K8: tiny cosine attention (n<=10 rows), scalar f32 — negligible FLOPs.
__global__ void k_cos_small(const float* __restrict__ X, float* __restrict__ Out, int n) {
    const int b = blockIdx.x, tid = threadIdx.x;
    const float* Xb = X + (size_t)b * n * 1024;
    float* Ob = Out + (size_t)b * n * 1024;
    __shared__ float red[256];
    __shared__ float invn[16];
    __shared__ float pm[16][16];
    for (int i = 0; i < n; ++i) {
        float ss = 0.f;
        for (int f = tid; f < 1024; f += 256) { float v = Xb[i * 1024 + f]; ss += v * v; }
        red[tid] = ss; __syncthreads();
        for (int s = 128; s > 0; s >>= 1) { if (tid < s) red[tid] += red[tid + s]; __syncthreads(); }
        if (tid == 0) invn[i] = rsqrtf(red[0]);
        __syncthreads();
    }
    if (tid < n * n) {
        int i = tid / n, j = tid % n;
        float d = 0.f;
        for (int f = 0; f < 1024; ++f) d += Xb[i * 1024 + f] * Xb[j * 1024 + f];
        pm[i][j] = d * invn[i] * invn[j];
    }
    __syncthreads();
    if (tid < n) {
        float m = -__builtin_inff();
        for (int j = 0; j < n; ++j) m = fmaxf(m, pm[tid][j]);
        float ssum = 0.f;
        for (int j = 0; j < n; ++j) { float e = __expf(pm[tid][j] - m); pm[tid][j] = e; ssum += e; }
        float inv = 1.f / ssum;
        for (int j = 0; j < n; ++j) pm[tid][j] *= inv;
    }
    __syncthreads();
    for (int e = tid; e < n * 1024; e += 256) {
        int i = e >> 10, f = e & 1023;
        float s = 0.f;
        for (int j = 0; j < n; ++j) s += pm[i][j] * Xb[j * 1024 + f];
        Ob[e] = s;
    }
}

// ---------------------------------------------------------------------------
extern "C" void kernel_launch(void* const* d_in, const int* in_sizes, int n_in,
                              void* d_out, int out_size, void* d_ws, size_t ws_size,
                              hipStream_t stream) {
    (void)in_sizes; (void)n_in; (void)out_size; (void)ws_size;
    const float* nor_feat  = (const float*)d_in[0];   // [64,512,1024]
    const float* feat_nor  = (const float*)d_in[1];   // [64,512,1024]
    const float* feat_abn  = (const float*)d_in[3];   // [64,512,1024]
    const float* score_abn = (const float*)d_in[4];   // [64,512]
    const float* feat_hard = (const float*)d_in[5];   // [64,1024,1024]
    const float* Wq = (const float*)d_in[7];  const float* bq = (const float*)d_in[8];
    const float* Wk = (const float*)d_in[9];  const float* bk = (const float*)d_in[10];

    float* out = (float*)d_out;
    float* o_new_score = out;                               // 64*1024
    float* o_sc_nor    = out + 65536;                       // 64*5
    float* o_sc_abn    = out + 65856;                       // 64*2
    float* o_nor_feat  = out + 65984;                       // 64*512*1024
    float* o_feat_nor  = out + 65984 + 33554432;            // 64*10*1024
    float* o_feat_abn  = o_feat_nor + 655360;               // 64*5*1024

    char* ws = (char*)d_ws;
    float*  feat_total = (float*)(ws);                      // 536,870,912 B
    __bf16* featn      = (__bf16*)(ws + 536870912ull);      // 268,435,456 B (feat+fh normalized)
    __bf16* xn         = (__bf16*)(ws + 805306368ull);      //  67,108,864 B
    float*  S          = (float*)(ws + 872415232ull);       //  67,108,864 B
    __bf16* P          = (__bf16*)(ws + 939524096ull);      //  33,554,432 B
    float*  ftop_nor   = (float*)(ws + 973078528ull);       //   2,621,440 B
    float*  ftop_abn   = (float*)(ws + 975699968ull);       //   1,310,720 B

    // 1) projections + bias + row-norm (bf16 copy)
    k_proj<<<dim3(32, 64), 128, 0, stream>>>(feat_nor, feat_abn, feat_hard,
                                             Wq, bq, Wk, bk, feat_total, featn);
    // 2) flash score attention -> new_score_hard
    k_attn_score<<<dim3(16, 64), 128, 0, stream>>>(featn, score_abn, o_new_score);
    // 3) normalize nor_feat_conf_nor
    k_rownorm<<<64 * 512, 256, 0, stream>>>(nor_feat, xn, 1024);
    // 4) S = xn @ xn^T
    k_gemm_nt_s<<<dim3(64, 64), 128, 0, stream>>>(xn, S);
    // 5) P = softmax(S)
    k_softmax_p<<<64 * 512, 256, 0, stream>>>(S, P);
    // 6) nor_new_feat_nor = P @ nor_feat
    k_gemm_pv<<<dim3(32, 16, 64), 128, 0, stream>>>(P, nor_feat, o_nor_feat);
    // 7) top-k + gathers
    k_topk_gather<<<64, 256, 0, stream>>>(score_abn, o_new_score, feat_total,
                                          ftop_nor, ftop_abn, o_sc_nor, o_sc_abn);
    // 8) tiny cosine attentions
    k_cos_small<<<64, 256, 0, stream>>>(ftop_nor, o_feat_nor, 10);
    k_cos_small<<<64, 256, 0, stream>>>(ftop_abn, o_feat_abn, 5);
}
